// Base_Q_40243843563642
// MI455X (gfx1250) — matrix operand order repaired and checked by reading, not compiled, not run on hardware
//
#include <hip/hip_runtime.h>
#include <hip/hip_bf16.h>

typedef __attribute__((ext_vector_type(16))) __bf16 v16bf;
typedef __attribute__((ext_vector_type(8)))  float  v8f;

#define N_D   256          // feature dim
#define K_CB  8192         // codebook rows
#define MT    128          // x rows per block
#define KC    64           // codebook rows per chunk
#define XLD   264          // x_lds row stride (bf16 elems), 528B -> 4-bank rotation/row
#define CBLD  264          // cb_lds row stride
#define CTLD  72           // cbt_lds / p_lds row stride (144B)
#define PLD   72

static __device__ __forceinline__ unsigned short bfbits(float f) {
  union { __bf16 h; unsigned short s; } u;
  u.h = (__bf16)f;           // RNE convert, v_cvt on gfx1250
  return u.s;
}
static __device__ __forceinline__ unsigned int pk2(float lo, float hi) {
  return (unsigned int)bfbits(lo) | ((unsigned int)bfbits(hi) << 16);
}

union FragBF { v16bf v; uint4 q[2]; };

__global__ __launch_bounds__(256) void vq_softq_kernel(
    const float* __restrict__ x, const float* __restrict__ cb,
    const int* __restrict__ soft_rate_p, float* __restrict__ out,
    int n_rows, int out_size)
{
  __shared__ alignas(16) __bf16 x_lds [MT * XLD];    // 67584 B
  __shared__ alignas(16) __bf16 cb_lds[KC * CBLD];   // 33792 B
  __shared__ alignas(16) __bf16 cbt_lds[N_D * CTLD]; // 36864 B
  __shared__ alignas(16) __bf16 p_lds [MT * PLD];    // 18432 B
  __shared__ float x2s[MT];
  __shared__ float c2s[KC];
  __shared__ float x2part[256];
  __shared__ float c2part[KC * 8];

  const int tid  = threadIdx.x;
  const int wave = tid >> 5;
  const int lane = tid & 31;
  const int hh   = lane >> 4;      // 16-lane half
  const int ln   = lane & 15;
  const int blk  = blockIdx.x;

  // ---------------- load x tile -> LDS bf16, compute ||x||^2 ----------------
  {
    const int r   = tid >> 1;
    const int cb0 = (tid & 1) * 128;
    const float4* src = (const float4*)(x + (size_t)(blk * MT + r) * N_D + cb0);
    float a2 = 0.f;
#pragma unroll
    for (int i = 0; i < 32; ++i) {
      float4 v = src[i];
      a2 = fmaf(v.x, v.x, fmaf(v.y, v.y, fmaf(v.z, v.z, fmaf(v.w, v.w, a2))));
      uint2 pk; pk.x = pk2(v.x, v.y); pk.y = pk2(v.z, v.w);
      *(uint2*)&x_lds[r * XLD + cb0 + 4 * i] = pk;
    }
    x2part[tid] = a2;
  }
  __syncthreads();
  if (tid < MT) x2s[tid] = x2part[2 * tid] + x2part[2 * tid + 1];
  __syncthreads();

  float x2j[8];
#pragma unroll
  for (int j = 0; j < 8; ++j) x2j[j] = x2s[16 * wave + j + 8 * hh];

  const float rate = (float)soft_rate_p[0];
  const float C1   = -rate * 1.4426950408889634f;   // -rate * log2(e): base-2 logits

  float mj[8], lj[8];
#pragma unroll
  for (int j = 0; j < 8; ++j) { mj[j] = -3.0e38f; lj[j] = 0.f; }
  v8f O[16];
#pragma unroll
  for (int t = 0; t < 16; ++t)
#pragma unroll
    for (int j = 0; j < 8; ++j) O[t][j] = 0.f;

  // ---------------- main loop over codebook chunks ----------------
  for (int ch = 0; ch < K_CB / KC; ++ch) {
    const int kc0 = ch * KC;

    // ---- stage codebook chunk: bf16 row-major + bf16 transposed + ||c||^2 ----
    {
      const int k0 = (tid >> 3) * 2;          // 2 codebook rows per thread
      const int db = (tid & 7) * 32;          // 32 D-columns per thread
      const float4* s0 = (const float4*)(cb + (size_t)(kc0 + k0) * N_D + db);
      const float4* s1 = (const float4*)(cb + (size_t)(kc0 + k0 + 1) * N_D + db);
      if (ch + 1 < K_CB / KC)                 // pull next slab toward L2/L0
        __builtin_prefetch(cb + (size_t)(kc0 + KC + k0) * N_D + db, 0, 0);
      float p0 = 0.f, p1 = 0.f;
#pragma unroll
      for (int i = 0; i < 8; ++i) {
        float4 a = s0[i];
        float4 b = s1[i];
        p0 = fmaf(a.x, a.x, fmaf(a.y, a.y, fmaf(a.z, a.z, fmaf(a.w, a.w, p0))));
        p1 = fmaf(b.x, b.x, fmaf(b.y, b.y, fmaf(b.z, b.z, fmaf(b.w, b.w, p1))));
        uint2 ra; ra.x = pk2(a.x, a.y); ra.y = pk2(a.z, a.w);
        uint2 rb; rb.x = pk2(b.x, b.y); rb.y = pk2(b.z, b.w);
        *(uint2*)&cb_lds[(k0    ) * CBLD + db + 4 * i] = ra;
        *(uint2*)&cb_lds[(k0 + 1) * CBLD + db + 4 * i] = rb;
        const int d = db + 4 * i;
        *(unsigned int*)&cbt_lds[(d + 0) * CTLD + k0] = pk2(a.x, b.x);
        *(unsigned int*)&cbt_lds[(d + 1) * CTLD + k0] = pk2(a.y, b.y);
        *(unsigned int*)&cbt_lds[(d + 2) * CTLD + k0] = pk2(a.z, b.z);
        *(unsigned int*)&cbt_lds[(d + 3) * CTLD + k0] = pk2(a.w, b.w);
      }
      c2part[(k0    ) * 8 + (tid & 7)] = p0;
      c2part[(k0 + 1) * 8 + (tid & 7)] = p1;
    }
    __syncthreads();
    if (tid < KC) {
      float s = 0.f;
#pragma unroll
      for (int i = 0; i < 8; ++i) s += c2part[tid * 8 + i];
      c2s[tid] = s;
    }
    __syncthreads();

    // ---- GEMM1: S[16 x 64] = Xw @ CBc^T via v_wmma_f32_16x16x32_bf16 ----
    v8f acc[4];
#pragma unroll
    for (int t = 0; t < 4; ++t)
#pragma unroll
      for (int j = 0; j < 8; ++j) acc[t][j] = 0.f;

#pragma unroll
    for (int kb = 0; kb < 8; ++kb) {
      FragBF A;
      const __bf16* ap = &x_lds[(16 * wave + ln) * XLD + kb * 32 + hh * 8];
      A.q[0] = *(const uint4*)ap;
      A.q[1] = *(const uint4*)(ap + 16);
#pragma unroll
      for (int t = 0; t < 4; ++t) {
        FragBF B;
        const __bf16* bp = &cb_lds[(16 * t + ln) * CBLD + kb * 32 + hh * 8];
        B.q[0] = *(const uint4*)bp;
        B.q[1] = *(const uint4*)(bp + 16);
        acc[t] = __builtin_amdgcn_wmma_f32_16x16x32_bf16(
            false, A.v, false, B.v, (short)0, acc[t], false, false);
      }
    }

    // ---- logits + in-wave online softmax (rows match C-layout halves) ----
    float c2v[4];
#pragma unroll
    for (int t = 0; t < 4; ++t) c2v[t] = c2s[16 * t + ln];
    float rm[8];
#pragma unroll
    for (int j = 0; j < 8; ++j) rm[j] = -3.0e38f;
#pragma unroll
    for (int t = 0; t < 4; ++t)
#pragma unroll
      for (int j = 0; j < 8; ++j) {
        float sq = fmaf(-2.f, acc[t][j], x2j[j] + c2v[t]);
        float s2 = C1 * sqrtf(fmaxf(sq, 0.f));     // log2-domain logit
        acc[t][j] = s2;
        rm[j] = fmaxf(rm[j], s2);
      }
#pragma unroll
    for (int j = 0; j < 8; ++j) {
      rm[j] = fmaxf(rm[j], __shfl_xor(rm[j], 1));
      rm[j] = fmaxf(rm[j], __shfl_xor(rm[j], 2));
      rm[j] = fmaxf(rm[j], __shfl_xor(rm[j], 4));
      rm[j] = fmaxf(rm[j], __shfl_xor(rm[j], 8));  // stays within 16-lane half
    }
    float alpha[8];
#pragma unroll
    for (int j = 0; j < 8; ++j) {
      float mn = fmaxf(mj[j], rm[j]);
      alpha[j] = exp2f(mj[j] - mn);
      mj[j] = mn;
    }
    float rs[8];
#pragma unroll
    for (int j = 0; j < 8; ++j) rs[j] = 0.f;
#pragma unroll
    for (int t = 0; t < 4; ++t)
#pragma unroll
      for (int j = 0; j < 8; ++j) {
        float p = exp2f(acc[t][j] - mj[j]);
        acc[t][j] = p;
        rs[j] += p;
      }
#pragma unroll
    for (int j = 0; j < 8; ++j) {
      rs[j] += __shfl_xor(rs[j], 1);
      rs[j] += __shfl_xor(rs[j], 2);
      rs[j] += __shfl_xor(rs[j], 4);
      rs[j] += __shfl_xor(rs[j], 8);
      lj[j] = fmaf(alpha[j], lj[j], rs[j]);
    }

    // ---- P -> LDS bf16 (wave-private rows; only DS wait needed) ----
#pragma unroll
    for (int t = 0; t < 4; ++t)
#pragma unroll
      for (int j = 0; j < 8; ++j)
        p_lds[(16 * wave + j + 8 * hh) * PLD + 16 * t + ln] = (__bf16)acc[t][j];

    // ---- GEMM2: O = diag(alpha)*O + P @ CBc ----
#pragma unroll
    for (int tt = 0; tt < 16; ++tt)
#pragma unroll
      for (int j = 0; j < 8; ++j) O[tt][j] *= alpha[j];
#pragma unroll
    for (int kb = 0; kb < 2; ++kb) {
      FragBF P;
      const __bf16* pp = &p_lds[(16 * wave + ln) * PLD + kb * 32 + hh * 8];
      P.q[0] = *(const uint4*)pp;
      P.q[1] = *(const uint4*)(pp + 16);
#pragma unroll
      for (int tt = 0; tt < 16; ++tt) {
        FragBF B;
        const __bf16* bp = &cbt_lds[(16 * tt + ln) * CTLD + kb * 32 + hh * 8];
        B.q[0] = *(const uint4*)bp;
        B.q[1] = *(const uint4*)(bp + 16);
        O[tt] = __builtin_amdgcn_wmma_f32_16x16x32_bf16(
            false, P.v, false, B.v, (short)0, O[tt], false, false);
      }
    }
    __syncthreads();   // protect cb_lds/cbt_lds before next chunk restage
  }

  // ---------------- epilogue: normalize and store ----------------
  float rl[8];
#pragma unroll
  for (int j = 0; j < 8; ++j) rl[j] = 1.0f / lj[j];
#pragma unroll
  for (int tt = 0; tt < 16; ++tt)
#pragma unroll
    for (int j = 0; j < 8; ++j) {
      size_t row = (size_t)blk * MT + 16 * wave + j + 8 * hh;
      out[row * N_D + 16 * tt + ln] = O[tt][j] * rl[j];
    }
  // second tuple element of the reference is the constant -1
  if (blk == 0 && tid == 0) {
    size_t qelems = (size_t)n_rows * N_D;
    if ((size_t)out_size > qelems) out[qelems] = -1.0f;
  }
}

extern "C" void kernel_launch(void* const* d_in, const int* in_sizes, int n_in,
                              void* d_out, int out_size, void* d_ws, size_t ws_size,
                              hipStream_t stream) {
  const float* x   = (const float*)d_in[0];
  const float* cbk = (const float*)d_in[1];
  const int*   sr  = (const int*)d_in[2];
  float* out = (float*)d_out;
  const int n_rows = in_sizes[0] / N_D;      // 32768
  dim3 grid(n_rows / MT), block(256);
  hipLaunchKernelGGL(vq_softq_kernel, grid, block, 0, stream,
                     x, cbk, sr, out, n_rows, out_size);
}
